// DenseChebConv_theta_nosum_10703058502325
// MI455X (gfx1250) — matrix operand order repaired and checked by reading, not compiled, not run on hardware
//
#include <hip/hip_runtime.h>

typedef __bf16 bf16_t;
typedef __attribute__((ext_vector_type(16))) __bf16 v16bf;
typedef __attribute__((ext_vector_type(8)))  __bf16 v8bf;
typedef __attribute__((ext_vector_type(4)))  __bf16 v4bf;
typedef __attribute__((ext_vector_type(8)))  float  v8f;

#define BB   4
#define NN   4096
#define FF   32
#define KORD 6
#define OO   128

// ---------------------------------------------------------------------------
// L (fp32 [B,N,N]) -> Lb (bf16 [B,N,N]).  One streaming pass; Lb (128 MB)
// then lives in the 192 MB L2 for the 5 recurrence matmuls.
// ---------------------------------------------------------------------------
__global__ void k_cvt_bf16(const float* __restrict__ src, bf16_t* __restrict__ dst, long n) {
    long stride = (long)gridDim.x * blockDim.x * 4;
    for (long i = ((long)blockIdx.x * blockDim.x + threadIdx.x) * 4; i < n; i += stride) {
        float4 v = *(const float4*)(src + i);
        v4bf o;
        o.x = (bf16_t)v.x; o.y = (bf16_t)v.y; o.z = (bf16_t)v.z; o.w = (bf16_t)v.w;
        *(v4bf*)(dst + i) = o;
    }
}

// ---------------------------------------------------------------------------
// x (fp32 [B,N,F]) -> xT (bf16 [B,F,N]) : WMMA-B-fragment-friendly transpose.
// ---------------------------------------------------------------------------
__global__ void k_xt_bf16(const float* __restrict__ x, bf16_t* __restrict__ xt) {
    int idx = blockIdx.x * blockDim.x + threadIdx.x;   // < BB*NN*FF
    int b   = idx / (NN * FF);
    int r   = idx - b * NN * FF;
    int row = r / FF;
    int f   = r - row * FF;
    xt[(size_t)b * FF * NN + (size_t)f * NN + row] = (bf16_t)x[idx];
}

// ---------------------------------------------------------------------------
// Tout = (use_prev2 ? 2*(Lb@Tprev) - Tprev2 : Lb@Tprev), per batch.
// Lb: bf16 [B,N,N] row-major (A matrix), TBprev: bf16 [B,F,N] (= T^T, B matrix),
// Tprev2/Tout: fp32 [B,N,F], TBout: bf16 [B,F,N].
// One wave per 16-row strip; 4 waves / block; 2 WMMA accumulators (F = 2x16).
// ---------------------------------------------------------------------------
__global__ __launch_bounds__(128)
void k_cheb_mm(const bf16_t* __restrict__ Lb, const bf16_t* __restrict__ TBprev,
               const float* __restrict__ Tprev2, float* __restrict__ Tout,
               bf16_t* __restrict__ TBout, int use_prev2)
{
    const int lane  = threadIdx.x & 31;
    const int wave  = threadIdx.x >> 5;
    const int tile  = blockIdx.x * 4 + wave;
    const int tpb   = NN / 16;                // tiles per batch
    const int b     = tile / tpb;
    const int mbase = (tile - b * tpb) * 16;
    const int hf    = lane >> 4;              // lane half (ISA A/B layouts)
    const int ln    = lane & 15;

    // A: lane handles row (mbase+ln); K chunks [8*hf .. 8*hf+7] and [16+8*hf .. +7]
    const bf16_t* Arow = Lb + (size_t)b * NN * NN + (size_t)(mbase + ln) * NN;
    // B: lane handles column f = (f0+ln); K chunks [8*hf .. 8*hf+7] and [16+8*hf .. +7]
    const bf16_t* TBb  = TBprev + (size_t)b * FF * NN;
    const bf16_t* tbp0 = TBb + (size_t)ln        * NN + 8 * hf;
    const bf16_t* tbp1 = TBb + (size_t)(16 + ln) * NN + 8 * hf;

    v8f acc0 = {}; v8f acc1 = {};
    #pragma unroll 4
    for (int kc = 0; kc < NN; kc += 32) {
        v8bf alo = *(const v8bf*)(Arow + kc + 8 * hf);
        v8bf ahi = *(const v8bf*)(Arow + kc + 16 + 8 * hf);
        v8bf b0a = *(const v8bf*)(tbp0 + kc);
        v8bf b0b = *(const v8bf*)(tbp0 + kc + 16);
        v8bf b1a = *(const v8bf*)(tbp1 + kc);
        v8bf b1b = *(const v8bf*)(tbp1 + kc + 16);
        v16bf a, b0, b1;
        #pragma unroll
        for (int e = 0; e < 8; ++e) {
            a[e]  = alo[e]; a[e + 8]  = ahi[e];
            b0[e] = b0a[e]; b0[e + 8] = b0b[e];
            b1[e] = b1a[e]; b1[e + 8] = b1b[e];
        }
        acc0 = __builtin_amdgcn_wmma_f32_16x16x32_bf16(false, a, false, b0, (short)0, acc0, false, false);
        acc1 = __builtin_amdgcn_wmma_f32_16x16x32_bf16(false, a, false, b1, (short)0, acc1, false, false);
    }

    // Epilogue. C/D layout: lane = column (ln), VGPR r -> row m = r + 8*hf.
    const size_t baseNF = (size_t)b * NN * FF;
    v8bf st0, st1;
    #pragma unroll
    for (int r = 0; r < 8; ++r) {
        const int row = mbase + r + 8 * hf;
        float v0 = acc0[r], v1 = acc1[r];
        if (use_prev2) {
            v0 = 2.0f * v0 - Tprev2[baseNF + (size_t)row * FF + ln];
            v1 = 2.0f * v1 - Tprev2[baseNF + (size_t)row * FF + 16 + ln];
        }
        Tout[baseNF + (size_t)row * FF + ln]      = v0;
        Tout[baseNF + (size_t)row * FF + 16 + ln] = v1;
        st0[r] = (bf16_t)v0;
        st1[r] = (bf16_t)v1;
    }
    // Transposed bf16 copy: rows contiguous -> one packed b128 store per tile.
    bf16_t* TBo = TBout + (size_t)b * FF * NN;
    *(v8bf*)(TBo + (size_t)ln        * NN + mbase + 8 * hf) = st0;
    *(v8bf*)(TBo + (size_t)(16 + ln) * NN + mbase + 8 * hf) = st1;
}

// ---------------------------------------------------------------------------
// out[b,rows,:] (+)= theta * (T[b,rows,:] @ Wk), T fp32 [B,N,F], Wk fp32 [F,O].
// One wave per 16-row strip; 8 output N-tiles over O=128; F=32 -> one K step.
// ---------------------------------------------------------------------------
__global__ __launch_bounds__(128)
void k_proj(const float* __restrict__ T, const float* __restrict__ Wk,
            const float* __restrict__ thetak, float* __restrict__ out, int accumulate)
{
    const int lane  = threadIdx.x & 31;
    const int wave  = threadIdx.x >> 5;
    const int tile  = blockIdx.x * 4 + wave;
    const int tpb   = NN / 16;
    const int b     = tile / tpb;
    const int mbase = (tile - b * tpb) * 16;
    const int hf    = lane >> 4;
    const int ln    = lane & 15;
    const float th  = *thetak;

    // A fragment from row-major fp32 T, converted in-register to bf16.
    const float* trow = T + (size_t)b * NN * FF + (size_t)(mbase + ln) * FF;
    float4 a0 = *(const float4*)(trow + 8 * hf);
    float4 a1 = *(const float4*)(trow + 8 * hf + 4);
    float4 a2 = *(const float4*)(trow + 16 + 8 * hf);
    float4 a3 = *(const float4*)(trow + 16 + 8 * hf + 4);
    v16bf a;
    a[0]  = (bf16_t)a0.x; a[1]  = (bf16_t)a0.y; a[2]  = (bf16_t)a0.z; a[3]  = (bf16_t)a0.w;
    a[4]  = (bf16_t)a1.x; a[5]  = (bf16_t)a1.y; a[6]  = (bf16_t)a1.z; a[7]  = (bf16_t)a1.w;
    a[8]  = (bf16_t)a2.x; a[9]  = (bf16_t)a2.y; a[10] = (bf16_t)a2.z; a[11] = (bf16_t)a2.w;
    a[12] = (bf16_t)a3.x; a[13] = (bf16_t)a3.y; a[14] = (bf16_t)a3.z; a[15] = (bf16_t)a3.w;

    float* outb = out + (size_t)b * NN * OO;
    #pragma unroll
    for (int t = 0; t < 8; ++t) {
        const int o0 = t * 16;
        const float* wcol = Wk + (size_t)(8 * hf) * OO + o0 + ln;   // W is 16 KB: L2/L0 hot
        v16bf bw;
        #pragma unroll
        for (int e = 0; e < 16; ++e) bw[e] = (bf16_t)wcol[(size_t)(e + (e & 8)) * OO];
        v8f acc = {};
        acc = __builtin_amdgcn_wmma_f32_16x16x32_bf16(false, a, false, bw, (short)0, acc, false, false);
        #pragma unroll
        for (int r = 0; r < 8; ++r) {
            const size_t idx = (size_t)(mbase + r + 8 * hf) * OO + o0 + ln;
            float v = th * acc[r];
            if (accumulate) v += outb[idx];
            outb[idx] = v;
        }
    }
}

// ---------------------------------------------------------------------------
extern "C" void kernel_launch(void* const* d_in, const int* in_sizes, int n_in,
                              void* d_out, int out_size, void* d_ws, size_t ws_size,
                              hipStream_t stream)
{
    (void)in_sizes; (void)n_in; (void)out_size; (void)ws_size;
    const float* x     = (const float*)d_in[0];
    const float* L     = (const float*)d_in[1];
    const float* W     = (const float*)d_in[2];
    const float* theta = (const float*)d_in[3];
    float* out = (float*)d_out;

    // Workspace carve-out (~144 MB): Lb 128MB, 3x fp32 T (2MB), 3x bf16 T^T (1MB)
    char* ws = (char*)d_ws;
    size_t off = 0;
    auto carve = [&](size_t bytes) -> void* {
        void* p = ws + off;
        off = (off + bytes + 255) & ~(size_t)255;
        return p;
    };
    bf16_t* Lb = (bf16_t*)carve((size_t)BB * NN * NN * sizeof(bf16_t));
    float*  Tf[3];
    for (int i = 0; i < 3; ++i) Tf[i] = (float*)carve((size_t)BB * NN * FF * sizeof(float));
    bf16_t* TBx = (bf16_t*)carve((size_t)BB * NN * FF * sizeof(bf16_t));
    bf16_t* TBp[2];
    for (int i = 0; i < 2; ++i) TBp[i] = (bf16_t*)carve((size_t)BB * NN * FF * sizeof(bf16_t));

    k_cvt_bf16<<<4096, 256, 0, stream>>>(L, Lb, (long)BB * NN * NN);
    k_xt_bf16<<<(BB * NN * FF) / 256, 256, 0, stream>>>(x, TBx);

    const int blocks = (BB * (NN / 16)) / 4;   // 256 blocks x 4 waves

    // k = 0: out = theta0 * (x @ W0)   (overwrites poisoned d_out)
    k_proj<<<blocks, 128, 0, stream>>>(x, W, theta, out, 0);

    for (int k = 1; k < KORD; ++k) {
        const bf16_t* tbprev = (k == 1) ? TBx : TBp[(k - 1) & 1];
        const float*  prev2  = (k <= 2) ? x : Tf[(k - 2) % 3];   // unused when k==1
        k_cheb_mm<<<blocks, 128, 0, stream>>>(Lb, tbprev, prev2,
                                              Tf[k % 3], TBp[k & 1], (k >= 2) ? 1 : 0);
        k_proj<<<blocks, 128, 0, stream>>>(Tf[k % 3], W + (size_t)k * FF * OO,
                                           theta + k, out, 1);
    }
}
